// Attention2d_75041668596345
// MI455X (gfx1250) — matrix operand order repaired and checked by reading, not compiled, not run on hardware
//
#include <hip/hip_runtime.h>
#include <hip/hip_bf16.h>

// ---------------------------------------------------------------------------
// Problem constants (from reference)
// ---------------------------------------------------------------------------
#define BB_  16      // batch
#define CC_  512     // channels
#define SS_  32      // spatial
#define HH_  8       // heads
#define DD_  64      // head dim
#define SEQ_ 1024    // S*S
#define INNER_ 512   // H*D
#define NPIX_ (BB_*SEQ_)      // 16384 global pixels
#define EPS_  1e-5f
#define SCALE_ 0.125f         // D^-0.5

typedef __bf16 v16bf __attribute__((ext_vector_type(16)));
typedef __bf16 v8bf  __attribute__((ext_vector_type(8)));
typedef float  v8f   __attribute__((ext_vector_type(8)));

static __device__ __forceinline__ __bf16 f2bf(float f) {
    unsigned u; __builtin_memcpy(&u, &f, 4);
    u += 0x7FFFu + ((u >> 16) & 1u);          // round-to-nearest-even
    unsigned short s = (unsigned short)(u >> 16);
    __bf16 b; __builtin_memcpy(&b, &s, 2);
    return b;
}
static __device__ __forceinline__ float bf2f(__bf16 b) {
    unsigned short s; __builtin_memcpy(&s, &b, 2);
    unsigned u = ((unsigned)s) << 16;
    float f; __builtin_memcpy(&f, &u, 4);
    return f;
}

// ---------------------------------------------------------------------------
// 0) fp32 -> bf16 weight conversion
// ---------------------------------------------------------------------------
__global__ void __launch_bounds__(256) cvt_bf16_kernel(const float* __restrict__ in,
                                                       __bf16* __restrict__ out, int n) {
    int i = blockIdx.x * 256 + threadIdx.x;
    if (i < n) out[i] = f2bf(in[i]);
}

// ---------------------------------------------------------------------------
// 1) scale-only channel LayerNorm; writes xn pixel-major [NPIX, C] bf16
// ---------------------------------------------------------------------------
__global__ void __launch_bounds__(256) ln_kernel(const float* __restrict__ x,
                                                 const float* __restrict__ scale,
                                                 __bf16* __restrict__ xn) {
    int t = blockIdx.x * 256 + threadIdx.x;       // global pixel, 0..16383
    int b = t >> 10, p = t & 1023;
    const float* xp = x + ((size_t)b << 19) + p;  // x[b, c, p], stride 1024 over c
    float s = 0.f, ss = 0.f;
    for (int c = 0; c < CC_; ++c) {
        float v = xp[(size_t)c << 10];
        s += v; ss += v * v;
    }
    float mean = s * (1.f / CC_);
    float var  = ss * (1.f / CC_) - mean * mean;
    float rs   = rsqrtf(var + EPS_);
    __bf16* op = xn + ((size_t)t << 9);
    for (int c = 0; c < CC_; ++c)
        op[c] = f2bf((xp[(size_t)c << 10] - mean) * rs * scale[c]);
}

// ---------------------------------------------------------------------------
// 2) WMMA GEMM: out[b, o, p] = sum_c A[o, c] * Bm[b*SEQ+p, c]
//    A: [OC, K] bf16 row-major, Bm: [NPIX, K] bf16 row-major
//    out: channel-major [(b*OC + o)*1024 + p], fp32 or bf16
//    wave computes 16(M) x 64(N); block = 8 waves; grid(OC/16, NPIX/512)
// ---------------------------------------------------------------------------
template <bool BF16OUT>
__global__ void __launch_bounds__(256) gemm_kernel(const __bf16* __restrict__ A,
                                                   const __bf16* __restrict__ Bm,
                                                   void* __restrict__ outp,
                                                   int OC, int K) {
    const int lane = threadIdx.x & 31;
    const int wave = threadIdx.x >> 5;
    const int l    = lane & 15;
    const int half = lane >> 4;
    const int mt   = blockIdx.x;
    const int n0   = (blockIdx.y * 8 + wave) * 64;

    const __bf16* Arow = A + (size_t)(mt * 16 + l) * K;
    const __bf16* B0 = Bm + (size_t)(n0 + l) * K + half * 16;
    const __bf16* B1 = B0 + (size_t)16 * K;
    const __bf16* B2 = B0 + (size_t)32 * K;
    const __bf16* B3 = B0 + (size_t)48 * K;

    v8f acc[4];
    for (int t = 0; t < 4; ++t) acc[t] = (v8f){};

    for (int k0 = 0; k0 < K; k0 += 32) {
        v8bf alo = *(const v8bf*)(Arow + k0 + half * 16);
        v8bf ahi = *(const v8bf*)(Arow + k0 + half * 16 + 8);
        v16bf a = __builtin_shufflevector(alo, ahi,
                    0,1,2,3,4,5,6,7,8,9,10,11,12,13,14,15);
        v16bf b0 = *(const v16bf*)(B0 + k0);
        v16bf b1 = *(const v16bf*)(B1 + k0);
        v16bf b2 = *(const v16bf*)(B2 + k0);
        v16bf b3 = *(const v16bf*)(B3 + k0);
        acc[0] = __builtin_amdgcn_wmma_f32_16x16x32_bf16(false, a, false, b0, (short)0, acc[0], false, false);
        acc[1] = __builtin_amdgcn_wmma_f32_16x16x32_bf16(false, a, false, b1, (short)0, acc[1], false, false);
        acc[2] = __builtin_amdgcn_wmma_f32_16x16x32_bf16(false, a, false, b2, (short)0, acc[2], false, false);
        acc[3] = __builtin_amdgcn_wmma_f32_16x16x32_bf16(false, a, false, b3, (short)0, acc[3], false, false);
    }

    #pragma unroll
    for (int t = 0; t < 4; ++t) {
        int gp = n0 + t * 16 + l;
        int bb = gp >> 10, p = gp & 1023;
        #pragma unroll
        for (int r = 0; r < 8; ++r) {
            size_t idx = ((size_t)(bb * OC + mt * 16 + 8 * half + r) << 10) + p;
            if (BF16OUT) ((__bf16*)outp)[idx] = f2bf(acc[t][r]);
            else         ((float*)outp)[idx]  = acc[t][r];
        }
    }
}

// ---------------------------------------------------------------------------
// 3) depthwise 3x3 conv on qkv [B,1536,32,32] bf16; writes
//    q,k -> [b,h,seq,64] bf16 (q scaled by D^-0.5), v -> [b,h,64,seq] bf16
// ---------------------------------------------------------------------------
__global__ void __launch_bounds__(256) dwconv_kernel(
    const __bf16* __restrict__ qkv,
    const float* __restrict__ wq, const float* __restrict__ bq,
    const float* __restrict__ wk, const float* __restrict__ bk,
    const float* __restrict__ wv, const float* __restrict__ bv,
    __bf16* __restrict__ qo, __bf16* __restrict__ ko, __bf16* __restrict__ vT) {
    int t = blockIdx.x * 256 + threadIdx.x;       // (b, o, p)
    int p = t & 1023;
    int rest = t >> 10;                           // b*1536 + o
    int o = rest % 1536;
    int b = rest / 1536;
    int which = o >> 9;                           // 0=q 1=k 2=v
    int c = o & 511;

    const float* w; const float* bias;
    if (which == 0)      { w = wq; bias = bq; }
    else if (which == 1) { w = wk; bias = bk; }
    else                 { w = wv; bias = bv; }

    int y = p >> 5, x = p & 31;
    const __bf16* in = qkv + ((size_t)rest << 10);
    const float* wc = w + c * 9;

    float s = 0.f;
    #pragma unroll
    for (int dy = -1; dy <= 1; ++dy) {
        int yy = y + dy;
        if (yy < 0 || yy >= SS_) continue;
        #pragma unroll
        for (int dx = -1; dx <= 1; ++dx) {
            int xx = x + dx;
            if (xx < 0 || xx >= SS_) continue;
            s += bf2f(in[yy * 32 + xx]) * wc[(dy + 1) * 3 + (dx + 1)];
        }
    }
    s += bias[c];

    int h = c >> 6, d = c & 63;
    if (which == 0) {
        s *= SCALE_;
        qo[(((size_t)(b * 8 + h) << 10) + p) * 64 + d] = f2bf(s);
    } else if (which == 1) {
        ko[(((size_t)(b * 8 + h) << 10) + p) * 64 + d] = f2bf(s);
    } else {
        vT[((size_t)(b * 8 + h) * 64 + d) * 1024 + p] = f2bf(s);
    }
}

// ---------------------------------------------------------------------------
// 4) expand relative-position bias into per-tile WMMA C/D fragment layout:
//    relbf[((h*64 + it)*64 + jt)*256 + lane*8 + r] =
//        pos_bias[pos_idx[(it*16 + r + 8*(lane>>4))*1024 + jt*16 + (lane&15)], h]
//    so each lane's 8 bias values for one 16x16 tile are a 16-byte load.
// ---------------------------------------------------------------------------
__global__ void __launch_bounds__(256) bias_expand_kernel(const float* __restrict__ pos_bias,
                                                          const int* __restrict__ pos_idx,
                                                          __bf16* __restrict__ relbf) {
    int t = blockIdx.x * 256 + threadIdx.x;       // 0 .. 8*64*64*256-1
    int r    = t & 7;
    int lane = (t >> 3) & 31;
    int jt   = (t >> 8) & 63;
    int it   = (t >> 14) & 63;
    int h    = t >> 20;
    int i = it * 16 + r + 8 * (lane >> 4);
    int j = jt * 16 + (lane & 15);
    relbf[t] = f2bf(pos_bias[pos_idx[(i << 10) + j] * HH_ + h]);
}

// ---------------------------------------------------------------------------
// 5) flash attention, one wave per 16 query rows of one (b,h)
//    grid(B*H, SEQ/128), block 256 (8 waves). Output att [b,seq,512] bf16.
// ---------------------------------------------------------------------------
__global__ void __launch_bounds__(256) flash_attn_kernel(
    const __bf16* __restrict__ q, const __bf16* __restrict__ k,
    const __bf16* __restrict__ vT, const __bf16* __restrict__ relbf,
    __bf16* __restrict__ att) {
    __shared__ __align__(16) __bf16 pl[8][512];   // per-wave 16x32 P staging

    const int lane = threadIdx.x & 31;
    const int wave = threadIdx.x >> 5;
    const int l    = lane & 15;
    const int half = lane >> 4;
    const int bh   = blockIdx.x;
    const int b    = bh >> 3, h = bh & 7;
    const int itile = blockIdx.y * 8 + wave;
    const int i0    = itile * 16;

    // Q A-fragments (K = d 0..31 and 32..63), loaded once
    const __bf16* Qb = q + (((size_t)bh << 10) + i0 + l) * 64;
    v8bf ql0 = *(const v8bf*)(Qb + half * 16);
    v8bf qh0 = *(const v8bf*)(Qb + half * 16 + 8);
    v16bf qa0 = __builtin_shufflevector(ql0, qh0, 0,1,2,3,4,5,6,7,8,9,10,11,12,13,14,15);
    v8bf ql1 = *(const v8bf*)(Qb + 32 + half * 16);
    v8bf qh1 = *(const v8bf*)(Qb + 40 + half * 16);
    v16bf qa1 = __builtin_shufflevector(ql1, qh1, 0,1,2,3,4,5,6,7,8,9,10,11,12,13,14,15);

    v8f acc[4];
    #pragma unroll
    for (int t = 0; t < 4; ++t) acc[t] = (v8f){};
    float m_r[8], l_r[8];
    #pragma unroll
    for (int r = 0; r < 8; ++r) { m_r[r] = -1e30f; l_r[r] = 0.f; }

    const __bf16* Kbase = k  + ((size_t)bh << 10) * 64;
    const __bf16* Vbase = vT + ((size_t)bh << 6) * 1024;
    // pre-swizzled bias fragments for this (h, itile): 64 j-tiles of 256
    const __bf16* Bfrag = relbf + (((size_t)h * 64 + itile) * 64) * 256 + lane * 8;

    for (int j0 = 0; j0 < SEQ_; j0 += 32) {
        // ---- S = Q K^T for two 16-column tiles (j0.., j0+16..) ----
        v8f sa = (v8f){}, sb = (v8f){};
        {
            const __bf16* Kp = Kbase + (size_t)(j0 + l) * 64 + half * 16;
            v16bf kb0 = *(const v16bf*)(Kp);
            v16bf kb1 = *(const v16bf*)(Kp + 32);
            sa = __builtin_amdgcn_wmma_f32_16x16x32_bf16(false, qa0, false, kb0, (short)0, sa, false, false);
            sa = __builtin_amdgcn_wmma_f32_16x16x32_bf16(false, qa1, false, kb1, (short)0, sa, false, false);
            Kp += (size_t)16 * 64;
            v16bf kb2 = *(const v16bf*)(Kp);
            v16bf kb3 = *(const v16bf*)(Kp + 32);
            sb = __builtin_amdgcn_wmma_f32_16x16x32_bf16(false, qa0, false, kb2, (short)0, sb, false, false);
            sb = __builtin_amdgcn_wmma_f32_16x16x32_bf16(false, qa1, false, kb3, (short)0, sb, false, false);
        }
        // ---- add relative position bias (fragment-layout, 2x16B loads) ----
        {
            const __bf16* bp = Bfrag + (size_t)(j0 >> 4) * 256;
            v8bf ba = *(const v8bf*)bp;
            v8bf bb2 = *(const v8bf*)(bp + 256);
            #pragma unroll
            for (int r = 0; r < 8; ++r) {
                sa[r] += bf2f(ba[r]);
                sb[r] += bf2f(bb2[r]);
            }
        }
        // ---- online softmax (rows live in 16-lane half-wave groups) ----
        #pragma unroll
        for (int r = 0; r < 8; ++r) {
            float mx = fmaxf(sa[r], sb[r]);
            mx = fmaxf(mx, __shfl_xor(mx, 1));
            mx = fmaxf(mx, __shfl_xor(mx, 2));
            mx = fmaxf(mx, __shfl_xor(mx, 4));
            mx = fmaxf(mx, __shfl_xor(mx, 8));
            float mo = m_r[r];
            float mn = fmaxf(mo, mx);
            m_r[r] = mn;
            float alpha = __expf(mo - mn);
            acc[0][r] *= alpha; acc[1][r] *= alpha;
            acc[2][r] *= alpha; acc[3][r] *= alpha;
            float pa = __expf(sa[r] - mn);
            float pb = __expf(sb[r] - mn);
            float rs = pa + pb;
            rs += __shfl_xor(rs, 1);
            rs += __shfl_xor(rs, 2);
            rs += __shfl_xor(rs, 4);
            rs += __shfl_xor(rs, 8);
            l_r[r] = l_r[r] * alpha + rs;
            pl[wave][(r + 8 * half) * 32 + l]      = f2bf(pa);
            pl[wave][(r + 8 * half) * 32 + 16 + l] = f2bf(pb);
        }
        // ---- reshape P (C/D layout) -> A-fragment via wave-private LDS ----
        v8bf plo = *(const v8bf*)&pl[wave][l * 32 + half * 16];
        v8bf phi = *(const v8bf*)&pl[wave][l * 32 + half * 16 + 8];
        v16bf pf = __builtin_shufflevector(plo, phi, 0,1,2,3,4,5,6,7,8,9,10,11,12,13,14,15);
        // ---- O += P V  (V columns are contiguous in vT) ----
        #pragma unroll
        for (int t2 = 0; t2 < 4; ++t2) {
            v16bf vb = *(const v16bf*)(Vbase + (size_t)(t2 * 16 + l) * 1024 + j0 + half * 16);
            acc[t2] = __builtin_amdgcn_wmma_f32_16x16x32_bf16(false, pf, false, vb, (short)0, acc[t2], false, false);
        }
    }

    // ---- normalize and store att[b, i, h*64 + d] ----
    #pragma unroll
    for (int r = 0; r < 8; ++r) {
        float inv = 1.0f / l_r[r];
        size_t row = ((size_t)(b * 1024 + i0 + r + 8 * half) << 9) + (h << 6) + l;
        #pragma unroll
        for (int t2 = 0; t2 < 4; ++t2)
            att[row + t2 * 16] = f2bf(acc[t2][r] * inv);
    }
}

// ---------------------------------------------------------------------------
// host-side launch
// ---------------------------------------------------------------------------
extern "C" void kernel_launch(void* const* d_in, const int* in_sizes, int n_in,
                              void* d_out, int out_size, void* d_ws, size_t ws_size,
                              hipStream_t stream) {
    const float* x        = (const float*)d_in[0];
    const float* scale    = (const float*)d_in[1];
    const float* w_qkv    = (const float*)d_in[2];
    const float* dw_w_q   = (const float*)d_in[3];
    const float* dw_b_q   = (const float*)d_in[4];
    const float* dw_w_k   = (const float*)d_in[5];
    const float* dw_b_k   = (const float*)d_in[6];
    const float* dw_w_v   = (const float*)d_in[7];
    const float* dw_b_v   = (const float*)d_in[8];
    const float* w_out    = (const float*)d_in[9];
    const float* pos_bias = (const float*)d_in[10];
    const int*   pos_idx  = (const int*)d_in[11];
    float* out = (float*)d_out;

    char* ws = (char*)d_ws;
    size_t off = 0;
    __bf16* xn     = (__bf16*)(ws + off); off += (size_t)NPIX_ * CC_ * 2;        // 16 MB
    __bf16* wqkv_b = (__bf16*)(ws + off); off += (size_t)3 * INNER_ * CC_ * 2;   // 1.5 MB
    __bf16* wout_b = (__bf16*)(ws + off); off += (size_t)CC_ * INNER_ * 2;       // 0.5 MB
    __bf16* qkv    = (__bf16*)(ws + off); off += (size_t)BB_ * 1536 * SEQ_ * 2;  // 48 MB
    __bf16* qarr   = (__bf16*)(ws + off); off += (size_t)BB_ * HH_ * SEQ_ * DD_ * 2;  // 16 MB
    __bf16* karr   = (__bf16*)(ws + off); off += (size_t)BB_ * HH_ * SEQ_ * DD_ * 2;  // 16 MB
    __bf16* vTarr  = (__bf16*)(ws + off); off += (size_t)BB_ * HH_ * DD_ * SEQ_ * 2;  // 16 MB
    __bf16* attb   = (__bf16*)(ws + off); off += (size_t)NPIX_ * INNER_ * 2;     // 16 MB
    __bf16* relbf  = (__bf16*)(ws + off); off += (size_t)HH_ * 64 * 64 * 256 * 2; // 16 MB

    // 0) weight conversions + swizzled bias table (independent of activations)
    cvt_bf16_kernel<<<(3 * INNER_ * CC_ + 255) / 256, 256, 0, stream>>>(w_qkv, wqkv_b, 3 * INNER_ * CC_);
    cvt_bf16_kernel<<<(CC_ * INNER_ + 255) / 256, 256, 0, stream>>>(w_out, wout_b, CC_ * INNER_);
    bias_expand_kernel<<<(HH_ * SEQ_ * SEQ_) / 256, 256, 0, stream>>>(pos_bias, pos_idx, relbf);

    // 1) layernorm -> xn [NPIX, C] bf16
    ln_kernel<<<NPIX_ / 256, 256, 0, stream>>>(x, scale, xn);

    // 2) qkv GEMM (M=1536, K=512, N=16384) -> qkv [B,1536,1024] bf16
    gemm_kernel<true><<<dim3(1536 / 16, NPIX_ / 512), 256, 0, stream>>>(wqkv_b, xn, qkv, 1536, CC_);

    // 3) depthwise 3x3 -> q, k, vT
    dwconv_kernel<<<(BB_ * 1536 * SEQ_) / 256, 256, 0, stream>>>(
        qkv, dw_w_q, dw_b_q, dw_w_k, dw_b_k, dw_w_v, dw_b_v, qarr, karr, vTarr);

    // 4) flash attention -> att [B, SEQ, 512] bf16
    flash_attn_kernel<<<dim3(BB_ * HH_, SEQ_ / 128), 256, 0, stream>>>(qarr, karr, vTarr, relbf, attb);

    // 5) out projection (M=512, K=512, N=16384) -> d_out fp32 [B,512,32,32]
    gemm_kernel<false><<<dim3(CC_ / 16, NPIX_ / 512), 256, 0, stream>>>(wout_b, attb, out, CC_, INNER_);

    (void)in_sizes; (void)n_in; (void)out_size; (void)ws_size;
}
